// MultiHeadSelfAttention_9526237462687
// MI455X (gfx1250) — matrix operand order repaired and checked by reading, not compiled, not run on hardware
//
#include <hip/hip_runtime.h>

// ---------------------------------------------------------------------------
// MI455X (gfx1250) multi-head self-attention.
// bf16 WMMA + TDM tensor loads (fallback: async LDS copies) + double buffering.
// B=4, T=2048, C=1024, H=16, D=64.  wave32, v_wmma_f32_16x16x32_bf16.
// ---------------------------------------------------------------------------

#define BB 4
#define TT 2048
#define CC 1024
#define HH 16
#define DD 64
#define C3 (3 * CC)

// fallback when TDM builtin is unavailable
#define USE_ASYNC_COPY 1

#if __has_builtin(__builtin_amdgcn_tensor_load_to_lds)
#define HAVE_TDM 1
#else
#define HAVE_TDM 0
#endif

typedef __attribute__((ext_vector_type(16))) __bf16 v16bf;
typedef __attribute__((ext_vector_type(8)))  float  v8f;

union Frag16 { v16bf v; uint4 u[2]; };           // 16 bf16 = 32B
union F8     { v8f   v; float f[8]; };           // 16x16 f32 C/D fragment
union U4S8   { uint4 u; unsigned short s[8]; };  // 8 bf16 bit view

__device__ __forceinline__ v8f wmma_bf16(v16bf a, v16bf b, v8f c) {
  return __builtin_amdgcn_wmma_f32_16x16x32_bf16(
      false, a, false, b, (short)0, c, false, false);
}

// ---------------------------------------------------------------------------
// Tensor Data Mover: one wave-level instruction DMAs a [rows x 64] bf16 tile
// (row stride = strideElems) from global to LDS.  D# packed per ISA 8.3/8.4:
// group0: count=1, lds_addr, global_addr, type=2; group1: data_size=2B,
// tensor dims == tile dims (no OOB clipping), tensor_dim0_stride.
// ---------------------------------------------------------------------------
#if HAVE_TDM
typedef __attribute__((ext_vector_type(4))) unsigned tdm_v4u;
typedef __attribute__((ext_vector_type(8))) int      tdm_v8i;
typedef __attribute__((ext_vector_type(4))) int      tdm_v4i;

__device__ __forceinline__ void tdm_load_tile64(unsigned lds_byte_off,
                                                const void* gptr,
                                                unsigned rows,
                                                unsigned strideElems) {
  const unsigned cols = 64;
  unsigned long long ga = (unsigned long long)gptr;
  tdm_v4u g0;
  g0[0] = 1u;                                            // count=1 (valid D#)
  g0[1] = lds_byte_off;                                  // lds_addr
  g0[2] = (unsigned)(ga & 0xffffffffu);                  // global_addr lo
  g0[3] = (unsigned)((ga >> 32) & 0x01ffffffu) | (2u << 30);  // hi | type=2
  tdm_v8i g1;
  g1[0] = 0x00010000;                                    // data_size = 2B
  g1[1] = (int)((cols & 0xffffu) << 16);                 // tensor_dim0 lo
  g1[2] = (int)((cols >> 16) | ((rows & 0xffffu) << 16));// dim0 hi | dim1 lo
  g1[3] = (int)((rows >> 16) | ((cols & 0xffffu) << 16));// dim1 hi | tile_dim0
  g1[4] = (int)(rows & 0xffffu);                         // tile_dim1 (dim2=0)
  g1[5] = (int)strideElems;                              // dim0_stride lo
  g1[6] = 0;                                             // stride hi
  g1[7] = 0;
  tdm_v4i g2 = {0, 0, 0, 0};
  tdm_v4i g3 = {0, 0, 0, 0};
#if defined(__clang_major__) && (__clang_major__ >= 23)
  tdm_v8i g4 = {0, 0, 0, 0, 0, 0, 0, 0};
  __builtin_amdgcn_tensor_load_to_lds(g0, g1, g2, g3, g4, 0);
#else
  __builtin_amdgcn_tensor_load_to_lds(g0, g1, g2, g3, 0);
#endif
}
#endif  // HAVE_TDM

#if USE_ASYNC_COPY
// per-lane 16B global -> LDS, tracked by ASYNCcnt (GVS addressing)
__device__ __forceinline__ void async_copy_b128(unsigned lds_byte_off,
                                                unsigned gmem_byte_off,
                                                unsigned long long sbase) {
  asm volatile("global_load_async_to_lds_b128 %0, %1, %2"
               :: "v"(lds_byte_off), "v"(gmem_byte_off), "s"(sbase)
               : "memory");
}
#endif

__device__ __forceinline__ void dma_wait_all() {
#if HAVE_TDM
#if __has_builtin(__builtin_amdgcn_s_wait_tensorcnt)
  __builtin_amdgcn_s_wait_tensorcnt(0);
#else
  asm volatile("s_wait_tensorcnt 0" ::: "memory");
#endif
#elif USE_ASYNC_COPY
  asm volatile("s_wait_asynccnt 0" ::: "memory");
#endif
}

// ---------------------------------------------------------------------------
// f32 -> bf16 conversion
// ---------------------------------------------------------------------------
struct BF4 { __bf16 x, y, z, w; };

__global__ void cvt_f32_to_bf16_kernel(const float* __restrict__ in,
                                       __bf16* __restrict__ out, int n4) {
  int i = blockIdx.x * blockDim.x + threadIdx.x;
  if (i < n4) {
    float4 v = ((const float4*)in)[i];
    BF4 o;
    o.x = (__bf16)v.x; o.y = (__bf16)v.y; o.z = (__bf16)v.z; o.w = (__bf16)v.w;
    ((BF4*)out)[i] = o;
  }
}

// ---------------------------------------------------------------------------
// GEMM: out[M,N] = A[M,K] * Bw[N,K]^T + bias[N]
// Block tile 128x128, 8 waves (4x2), wave tile 32x64, K staged 64 through
// double-buffered LDS (2 x 32KB).  16 WMMA per wave per stage; the next
// stage's TDM/async DMA overlaps the current stage's WMMAs.
// ---------------------------------------------------------------------------
#define GEMM_BUF_ELEMS (128 * 64 + 128 * 64)  // A tile | B tile per buffer

template <bool OUT_BF16>
__global__ void __launch_bounds__(256)
gemm_bf16_wmma_kernel(const __bf16* __restrict__ A,
                      const __bf16* __restrict__ Bw,
                      const float* __restrict__ bias,
                      void* __restrict__ Cout,
                      int M, int N, int K) {
  __shared__ __align__(16) __bf16 smem[2 * GEMM_BUF_ELEMS];  // 64KB

  const int tid  = threadIdx.x;
  const int wave = tid >> 5;
  const int lane = tid & 31;
  const int g    = lane >> 4;
  const int l15  = lane & 15;

  const int m0 = blockIdx.y * 128;
  const int n0 = blockIdx.x * 128;
  const int wm = (wave >> 1) * 32;  // 0/32/64/96
  const int wn = (wave & 1) * 64;   // 0/64

  F8 c[2][4] = {};

#if !HAVE_TDM && USE_ASYNC_COPY
  const unsigned long long aBase = (unsigned long long)(const void*)A;
  const unsigned long long bBase = (unsigned long long)(const void*)Bw;
#endif

  // stage tile k0 into buffer `buf`
  auto stage = [&](int k0, int buf) {
    const unsigned aOff = (unsigned)(buf * GEMM_BUF_ELEMS) * 2u;
    const unsigned bOff = aOff + 128u * 64u * 2u;
#if HAVE_TDM
    if (wave == 0)
      tdm_load_tile64(aOff, A + (size_t)m0 * K + k0, 128, (unsigned)K);
    else if (wave == 1)
      tdm_load_tile64(bOff, Bw + (size_t)n0 * K + k0, 128, (unsigned)K);
#elif USE_ASYNC_COPY
#pragma unroll
    for (int q = 0; q < 4; ++q) {
      int ch  = tid * 4 + q;
      int row = ch >> 3, col = (ch & 7) * 8;
      async_copy_b128(aOff + (unsigned)(row * 64 + col) * 2,
                      (unsigned)(((size_t)(m0 + row) * K + k0 + col) * 2),
                      aBase);
      async_copy_b128(bOff + (unsigned)(row * 64 + col) * 2,
                      (unsigned)(((size_t)(n0 + row) * K + k0 + col) * 2),
                      bBase);
    }
#else
    __bf16* lA = smem + buf * GEMM_BUF_ELEMS;
    __bf16* lB = lA + 128 * 64;
#pragma unroll
    for (int q = 0; q < 4; ++q) {
      int ch  = tid * 4 + q;
      int row = ch >> 3, col = (ch & 7) * 8;
      *(uint4*)&lA[row * 64 + col] =
          *(const uint4*)&A[(size_t)(m0 + row) * K + k0 + col];
      *(uint4*)&lB[row * 64 + col] =
          *(const uint4*)&Bw[(size_t)(n0 + row) * K + k0 + col];
    }
#endif
  };

  stage(0, 0);
  int buf = 0;

  for (int k0 = 0; k0 < K; k0 += 64) {
    dma_wait_all();        // current buffer's DMA complete (this wave)
    __syncthreads();       // visible to all waves
    if (k0 + 64 < K) stage(k0 + 64, buf ^ 1);  // overlap next DMA w/ compute

    const __bf16* lA = smem + buf * GEMM_BUF_ELEMS;
    const __bf16* lB = lA + 128 * 64;

#pragma unroll
    for (int ks = 0; ks < 2; ++ks) {
      const int kb = ks * 32;
      Frag16 a[2], b[4];
#pragma unroll
      for (int i = 0; i < 2; ++i) {
        int r = wm + i * 16 + l15;
        a[i].u[0] = *(const uint4*)&lA[r * 64 + kb + g * 8];
        a[i].u[1] = *(const uint4*)&lA[r * 64 + kb + g * 8 + 16];
      }
#pragma unroll
      for (int j = 0; j < 4; ++j) {
        int rn = wn + j * 16 + l15;
        b[j].u[0] = *(const uint4*)&lB[rn * 64 + kb + g * 8];
        b[j].u[1] = *(const uint4*)&lB[rn * 64 + kb + g * 8 + 16];
      }
#pragma unroll
      for (int i = 0; i < 2; ++i)
#pragma unroll
        for (int j = 0; j < 4; ++j)
          c[i][j].v = wmma_bf16(a[i].v, b[j].v, c[i][j].v);
    }
    buf ^= 1;
  }

  // epilogue: vgpr r holds row (r + 8*g), col = l15
  const int hi = g * 8;
#pragma unroll
  for (int i = 0; i < 2; ++i)
#pragma unroll
    for (int j = 0; j < 4; ++j) {
      int n = n0 + wn + j * 16 + l15;
      float bv = bias[n];
#pragma unroll
      for (int r = 0; r < 8; ++r) {
        int m = m0 + wm + i * 16 + hi + r;
        float val = c[i][j].f[r] + bv;
        if (OUT_BF16)
          ((__bf16*)Cout)[(size_t)m * N + n] = (__bf16)val;
        else
          ((float*)Cout)[(size_t)m * N + n] = val;
      }
    }
}

// ---------------------------------------------------------------------------
// Flash attention: one block per (b, h, 64 q-rows).  4 waves x 16 q-rows.
// 64-key tiles: S = Q*K^T (8 wmma), online softmax, O += P*V (8 wmma).
// K tile staged by TDM (wave 0) or async copies; V^T by packed transpose.
// ---------------------------------------------------------------------------
__global__ void __launch_bounds__(128)
attn_flash_wmma_kernel(const __bf16* __restrict__ qkv,  // [B,T,3C] bf16
                       __bf16* __restrict__ attnb) {    // [B,T,C]  bf16
  // lK [64 key][64 d] | lVT [64 d][64 key] | lP 4 waves x [16 q][64 key]
  __shared__ __align__(16) __bf16 smem[64 * 64 + 64 * 64 + 4 * 16 * 64];
  __bf16* lK  = smem;
  __bf16* lVT = smem + 64 * 64;
  __bf16* lP  = smem + 2 * 64 * 64;

  const int b    = blockIdx.x / HH;
  const int h    = blockIdx.x % HH;
  const int tid  = threadIdx.x;
  const int wave = tid >> 5;
  const int lane = tid & 31;
  const int g    = lane >> 4;
  const int l15  = lane & 15;
  const int q0   = blockIdx.y * 64 + wave * 16;

  const size_t base = (size_t)b * TT * C3;
#if !HAVE_TDM && USE_ASYNC_COPY
  const unsigned long long qkvBase = (unsigned long long)(const void*)qkv;
#endif

  // Q fragments: 16 rows x D=64 -> two 32-wide K chunks
  Frag16 qa[2];
  {
    size_t qrow = base + (size_t)(q0 + l15) * C3 + h * DD;
    qa[0].u[0] = *(const uint4*)&qkv[qrow + g * 8];
    qa[0].u[1] = *(const uint4*)&qkv[qrow + g * 8 + 16];
    qa[1].u[0] = *(const uint4*)&qkv[qrow + 32 + g * 8];
    qa[1].u[1] = *(const uint4*)&qkv[qrow + 32 + g * 8 + 16];
  }

  F8 o[4] = {};
  float rm[8], rl[8];
#pragma unroll
  for (int i = 0; i < 8; ++i) { rm[i] = -1e30f; rl[i] = 0.f; }
  const float sm_scale = 0.125f;  // 1/sqrt(64)

  for (int kc = 0; kc < TT; kc += 64) {
    __syncthreads();
    // stage K tile: 64 keys x 64 d (row stride 3C)
#if HAVE_TDM
    if (wave == 0)
      tdm_load_tile64(0u, qkv + base + (size_t)kc * C3 + CC + h * DD, 64, C3);
#elif USE_ASYNC_COPY
#pragma unroll
    for (int q = 0; q < 4; ++q) {
      int ch  = tid * 4 + q;
      int row = ch >> 3, col = (ch & 7) * 8;
      async_copy_b128(
          (unsigned)(row * 64 + col) * 2,
          (unsigned)((base + (size_t)(kc + row) * C3 + CC + h * DD + col) * 2),
          qkvBase);
    }
#else
#pragma unroll
    for (int q = 0; q < 4; ++q) {
      int ch  = tid * 4 + q;
      int row = ch >> 3, col = (ch & 7) * 8;
      *(uint4*)&lK[row * 64 + col] =
          *(const uint4*)&qkv[base + (size_t)(kc + row) * C3 + CC + h * DD + col];
    }
#endif
    // stage V^T with paired keys: 2 x 16B loads -> 8 packed b32 LDS stores
#pragma unroll
    for (int q = 0; q < 2; ++q) {
      int task = tid * 2 + q;          // 32 key-pairs x 8 d-chunks
      int kp = task >> 3, cd = (task & 7) * 8;
      U4S8 v0, v1;
      v0.u = *(const uint4*)&qkv[base + (size_t)(kc + 2 * kp) * C3 + 2 * CC + h * DD + cd];
      v1.u = *(const uint4*)&qkv[base + (size_t)(kc + 2 * kp + 1) * C3 + 2 * CC + h * DD + cd];
#pragma unroll
      for (int j = 0; j < 8; ++j) {
        unsigned pk = (unsigned)v0.s[j] | ((unsigned)v1.s[j] << 16);
        *(unsigned*)&lVT[(cd + j) * 64 + 2 * kp] = pk;
      }
    }
    dma_wait_all();
    __syncthreads();

    // S = Q * K^T : 16 q-rows x 64 keys, 4 n-tiles x 2 k-chunks = 8 wmma
    F8 s[4];
#pragma unroll
    for (int nh = 0; nh < 4; ++nh) {
      s[nh].v = v8f{};
      Frag16 kb;
      int kr = nh * 16 + l15;
      kb.u[0] = *(const uint4*)&lK[kr * 64 + g * 8];
      kb.u[1] = *(const uint4*)&lK[kr * 64 + g * 8 + 16];
      s[nh].v = wmma_bf16(qa[0].v, kb.v, s[nh].v);
      kb.u[0] = *(const uint4*)&lK[kr * 64 + 32 + g * 8];
      kb.u[1] = *(const uint4*)&lK[kr * 64 + 32 + g * 8 + 16];
      s[nh].v = wmma_bf16(qa[1].v, kb.v, s[nh].v);
    }

    // online softmax in C-fragment layout (vgpr i <-> row i + 8*g,
    // lanes <-> columns; xor masks 1/2/4/8 reduce each 16-lane half)
    float p[4][8];
#pragma unroll
    for (int i = 0; i < 8; ++i) {
      float e0 = s[0].f[i] * sm_scale, e1 = s[1].f[i] * sm_scale;
      float e2 = s[2].f[i] * sm_scale, e3 = s[3].f[i] * sm_scale;
      float mx = fmaxf(fmaxf(e0, e1), fmaxf(e2, e3));
#pragma unroll
      for (int off = 1; off < 16; off <<= 1)
        mx = fmaxf(mx, __shfl_xor(mx, off, 32));
      float nm  = fmaxf(rm[i], mx);
      float scl = __expf(rm[i] - nm);
      p[0][i] = __expf(e0 - nm);
      p[1][i] = __expf(e1 - nm);
      p[2][i] = __expf(e2 - nm);
      p[3][i] = __expf(e3 - nm);
      float sum = (p[0][i] + p[1][i]) + (p[2][i] + p[3][i]);
#pragma unroll
      for (int off = 1; off < 16; off <<= 1)
        sum += __shfl_xor(sum, off, 32);
      rl[i] = rl[i] * scl + sum;
      rm[i] = nm;
#pragma unroll
      for (int j = 0; j < 4; ++j) o[j].f[i] *= scl;
    }

    // bounce P (C layout) -> per-wave LDS patch -> A-fragment layout
    __bf16* pw = &lP[wave * (16 * 64)];
#pragma unroll
    for (int i = 0; i < 8; ++i) {
      int r = i + g * 8;
#pragma unroll
      for (int nh = 0; nh < 4; ++nh)
        pw[r * 64 + nh * 16 + l15] = (__bf16)p[nh][i];
    }
    // wave-local LDS RAW: split dep-counter wait, no block barrier needed
    asm volatile("s_wait_dscnt 0" ::: "memory");
    Frag16 pa[2];
#pragma unroll
    for (int kc2 = 0; kc2 < 2; ++kc2) {
      pa[kc2].u[0] = *(const uint4*)&pw[l15 * 64 + kc2 * 32 + g * 8];
      pa[kc2].u[1] = *(const uint4*)&pw[l15 * 64 + kc2 * 32 + g * 8 + 16];
    }

    // O += P(16x64) * V(64x64): 4 d-tiles x 2 k-chunks = 8 wmma
#pragma unroll
    for (int j = 0; j < 4; ++j) {
      int d = j * 16 + l15;
#pragma unroll
      for (int kc2 = 0; kc2 < 2; ++kc2) {
        Frag16 vb;
        vb.u[0] = *(const uint4*)&lVT[d * 64 + kc2 * 32 + g * 8];
        vb.u[1] = *(const uint4*)&lVT[d * 64 + kc2 * 32 + g * 8 + 16];
        o[j].v = wmma_bf16(pa[kc2].v, vb.v, o[j].v);
      }
    }
  }

  // normalize and store O as bf16 -> attn workspace [B,T,C]
#pragma unroll
  for (int j = 0; j < 4; ++j) {
    int col = h * DD + j * 16 + l15;
#pragma unroll
    for (int i = 0; i < 8; ++i) {
      int m = q0 + i + g * 8;
      attnb[((size_t)b * TT + m) * CC + col] = (__bf16)(o[j].f[i] / rl[i]);
    }
  }
}

// ---------------------------------------------------------------------------
// launch
// ---------------------------------------------------------------------------
extern "C" void kernel_launch(void* const* d_in, const int* in_sizes, int n_in,
                              void* d_out, int out_size, void* d_ws, size_t ws_size,
                              hipStream_t stream) {
  const float* x     = (const float*)d_in[0];
  const float* qkv_w = (const float*)d_in[1];
  const float* qkv_b = (const float*)d_in[2];
  const float* out_w = (const float*)d_in[3];
  const float* out_b = (const float*)d_in[4];
  float* out = (float*)d_out;

  const size_t nX    = (size_t)BB * TT * CC;   //  8,388,608
  const size_t nWqkv = (size_t)C3 * CC;        //  3,145,728
  const size_t nWo   = (size_t)CC * CC;        //  1,048,576
  const size_t nQKV  = (size_t)BB * TT * C3;   // 25,165,824

  __bf16* xb    = (__bf16*)d_ws;
  __bf16* wqkvb = xb + nX;
  __bf16* wob   = wqkvb + nWqkv;
  __bf16* qkvb  = wob + nWo;
  __bf16* attnb = qkvb + nQKV;

  // 1) bf16 conversions
  cvt_f32_to_bf16_kernel<<<(int)((nX / 4 + 255) / 256), 256, 0, stream>>>(
      x, xb, (int)(nX / 4));
  cvt_f32_to_bf16_kernel<<<(int)((nWqkv / 4 + 255) / 256), 256, 0, stream>>>(
      qkv_w, wqkvb, (int)(nWqkv / 4));
  cvt_f32_to_bf16_kernel<<<(int)((nWo / 4 + 255) / 256), 256, 0, stream>>>(
      out_w, wob, (int)(nWo / 4));

  // 2) qkv = x @ Wqkv^T + b : M=8192, N=3072, K=1024 -> bf16
  gemm_bf16_wmma_kernel<true>
      <<<dim3(3072 / 128, 8192 / 128), 256, 0, stream>>>(
          xb, wqkvb, qkv_b, (void*)qkvb, BB * TT, C3, CC);

  // 3) flash attention -> attn bf16 [B,T,C]
  attn_flash_wmma_kernel<<<dim3(BB * HH, TT / 64), 128, 0, stream>>>(qkvb, attnb);

  // 4) out = attn @ Wo^T + b : M=8192, N=1024, K=1024 -> f32
  gemm_bf16_wmma_kernel<false>
      <<<dim3(1024 / 128, 8192 / 128), 256, 0, stream>>>(
          attnb, wob, out_b, (void*)out, BB * TT, CC, CC);
}
